// PerformerSelfAttention_49976239456733
// MI455X (gfx1250) — matrix operand order repaired and checked by reading, not compiled, not run on hardware
//
#include <hip/hip_runtime.h>
#include <hip/hip_bf16.h>

// ---------------------------------------------------------------------------
// Performer self-attention, MI455X (gfx1250) wave32 + WMMA bf16 path.
// B=4 S=4096 D=1024 H=16 HD=64  -> rows = 16384
//
// Round 4: round-3 structure (8 waves x 2 heads, MT=4, unroll-1 outer loops,
// <=256 VGPRs), plus: batch the MT A-fragment LDS loads into a distinct
// af[MT] array ahead of the WMMA group in each K-step. Round 3's per-mt
// load/use interleave made the allocator reuse A-fragment registers right
// after the consuming v_wmma, forcing 4-wide WAR-hazard v_nop bursts in the
// hot loop; separating the load phase from the WMMA phase pushes the reuse
// distance past the hazard window.
// ---------------------------------------------------------------------------

typedef __attribute__((ext_vector_type(16))) __bf16 bf16x16;
typedef __attribute__((ext_vector_type(8)))  __bf16 bf16x8;
typedef __attribute__((ext_vector_type(8)))  float  f32x8;

#define D_DIM   1024
#define HEADS   16
#define HD      64
#define ROWS    16384          // B*S
#define MT      4              // 16-row M tiles per workgroup (64 rows)
#define KSTEPS  (D_DIM / 32)   // 32 wmma K-steps of 32
#define LDS_STRIDE (D_DIM + 16) // padded bf16 elements per LDS row

static __device__ __forceinline__ __bf16 f2bf(float f) {
  unsigned u = __builtin_bit_cast(unsigned, f);
  u += 0x7FFFu + ((u >> 16) & 1u);          // round-to-nearest-even
  unsigned short s = (unsigned short)(u >> 16);
  return __builtin_bit_cast(__bf16, s);
}

// ---------------- conversion kernels ----------------
__global__ void cvt_x_kernel(const float* __restrict__ x,
                             __bf16* __restrict__ xb, int n) {
  int i = blockIdx.x * blockDim.x + threadIdx.x;
  if (i < n) xb[i] = f2bf(x[i]);
}

// w[k][n] (row-major KxN) -> wt[n][k] (row-major NxK), bf16
__global__ void cvt_wT_kernel(const float* __restrict__ w,
                              __bf16* __restrict__ wt) {
  int i = blockIdx.x * blockDim.x + threadIdx.x;   // over 1M, coalesced read
  int k = i >> 10;
  int n = i & 1023;
  wt[(size_t)n * D_DIM + k] = f2bf(w[i]);
}

// ---------------- fragment helpers ----------------
static __device__ __forceinline__ bf16x16 load_a_frag(const __bf16* arow) {
  // A 16x32 fragment: per lane 16 contiguous k (k = 16*half + s), matching B
  bf16x8 lo = *(const bf16x8*)(arow);
  bf16x8 hi = *(const bf16x8*)(arow + 8);
  return __builtin_shufflevector(lo, hi,
                                 0, 1, 2, 3, 4, 5, 6, 7,
                                 8, 9, 10, 11, 12, 13, 14, 15);
}

static __device__ __forceinline__ f32x8 wmma_bf16(bf16x16 a, bf16x16 b, f32x8 c) {
  return __builtin_amdgcn_wmma_f32_16x16x32_bf16(false, a, false, b,
                                                 (short)0, c, false, false);
}

// ---------------- fused QKV + feature map + coeff*v ----------------
__global__ void __launch_bounds__(256)
qkv_kernel(const __bf16* __restrict__ xb,
           const __bf16* __restrict__ wqT,
           const __bf16* __restrict__ wkT,
           const __bf16* __restrict__ wvT,
           const float*  __restrict__ bq,
           const float*  __restrict__ bk,
           const float*  __restrict__ bv,
           __bf16* __restrict__ qkv)
{
  extern __shared__ __bf16 smem[];           // [MT*16][LDS_STRIDE]
  const int tid     = threadIdx.x;
  const int wave    = tid >> 5;              // 0..7
  const int lane    = tid & 31;
  const int lane_lo = lane & 15;
  const int half    = lane >> 4;
  const int mrow0   = blockIdx.x * (MT * 16);

  // cooperative stage of 64x1024 bf16 x-tile into LDS
  {
    const uint4* g = (const uint4*)(xb + (size_t)mrow0 * D_DIM);
    for (int i = tid; i < MT * 16 * (D_DIM / 8); i += 256) {
      int r = i >> 7;                        // D_DIM/8 == 128 chunks per row
      int c = i & 127;
      *(uint4*)(smem + r * LDS_STRIDE + c * 8) = g[r * 128 + c];
    }
  }
  __syncthreads();

  // per-lane LDS base for A fragments (row lane_lo of each M tile)
  const __bf16* abase = smem + lane_lo * LDS_STRIDE + half * 16;

  // each wave handles two heads sequentially: head = wave, wave + 8
#pragma unroll 1
  for (int hp = 0; hp < 2; ++hp) {
    const int c0 = (wave + hp * 8) * HD;     // this head's column base

    f32x8 cf[MT];                            // per-lane partial sum of q*k
#pragma unroll
    for (int mt = 0; mt < MT; ++mt)
#pragma unroll
      for (int j = 0; j < 8; ++j) cf[mt][j] = 0.f;

    // ---------- Q and K phase (fused K-loop, shared A fragments) ----------
#pragma unroll 1
    for (int nt = 0; nt < 4; ++nt) {
      const int colg = c0 + nt * 16 + lane_lo;
      const __bf16* wq_l = wqT + (size_t)colg * D_DIM + half * 16;
      const __bf16* wk_l = wkT + (size_t)colg * D_DIM + half * 16;

      f32x8 cq[MT], ck[MT];
#pragma unroll
      for (int mt = 0; mt < MT; ++mt)
#pragma unroll
        for (int j = 0; j < 8; ++j) { cq[mt][j] = 0.f; ck[mt][j] = 0.f; }

      for (int ks = 0; ks < KSTEPS; ++ks) {
        const int k0 = ks * 32;
        bf16x16 bqf = *(const bf16x16*)(wq_l + k0);
        bf16x16 bkf = *(const bf16x16*)(wk_l + k0);
        bf16x16 af[MT];
#pragma unroll
        for (int mt = 0; mt < MT; ++mt)
          af[mt] = load_a_frag(abase + mt * 16 * LDS_STRIDE + k0);
#pragma unroll
        for (int mt = 0; mt < MT; ++mt) {
          cq[mt] = wmma_bf16(af[mt], bqf, cq[mt]);
          ck[mt] = wmma_bf16(af[mt], bkf, ck[mt]);
        }
      }

      const float bqv = bq[colg];
      const float bkv = bk[colg];
#pragma unroll
      for (int mt = 0; mt < MT; ++mt)
#pragma unroll
        for (int j = 0; j < 8; ++j) {
          float tq = cq[mt][j] + bqv;
          float tk = ck[mt][j] + bkv;
          float fq = __expf(-0.5f * tq * tq);
          float fk = __expf(-0.5f * tk * tk);
          cf[mt][j] += fq * fk;
        }
    }

    // ---------- reduce coeff over the head dim (16 lanes per half) ----------
#pragma unroll
    for (int mt = 0; mt < MT; ++mt)
#pragma unroll
      for (int j = 0; j < 8; ++j) {
        float v = cf[mt][j];
        v += __shfl_xor(v, 1, 32);
        v += __shfl_xor(v, 2, 32);
        v += __shfl_xor(v, 4, 32);
        v += __shfl_xor(v, 8, 32);
        cf[mt][j] = v;   // coeff for row (mt*16 + j + 8*half)
      }

    // ---------- V phase: qkv = coeff * (x@Wv + bv) ----------
#pragma unroll 1
    for (int nt = 0; nt < 4; ++nt) {
      const int colg = c0 + nt * 16 + lane_lo;
      const __bf16* wv_l = wvT + (size_t)colg * D_DIM + half * 16;

      f32x8 cv[MT];
#pragma unroll
      for (int mt = 0; mt < MT; ++mt)
#pragma unroll
        for (int j = 0; j < 8; ++j) cv[mt][j] = 0.f;

      for (int ks = 0; ks < KSTEPS; ++ks) {
        const int k0 = ks * 32;
        bf16x16 bvf = *(const bf16x16*)(wv_l + k0);
        bf16x16 af[MT];
#pragma unroll
        for (int mt = 0; mt < MT; ++mt)
          af[mt] = load_a_frag(abase + mt * 16 * LDS_STRIDE + k0);
#pragma unroll
        for (int mt = 0; mt < MT; ++mt)
          cv[mt] = wmma_bf16(af[mt], bvf, cv[mt]);
      }

      const float bvv = bv[colg];
#pragma unroll
      for (int mt = 0; mt < MT; ++mt)
#pragma unroll
        for (int j = 0; j < 8; ++j) {
          float val = cf[mt][j] * (cv[mt][j] + bvv);
          int rowg = mrow0 + mt * 16 + j + half * 8;
          qkv[(size_t)rowg * D_DIM + colg] = f2bf(val);
        }
    }
  }
}

// ---------------- out = qkv @ Wo + bo ----------------
__global__ void __launch_bounds__(256)
out_kernel(const __bf16* __restrict__ qkv,
           const __bf16* __restrict__ woT,
           const float*  __restrict__ bo,
           float* __restrict__ out)
{
  extern __shared__ __bf16 smem[];           // [MT*16][LDS_STRIDE]
  const int tid     = threadIdx.x;
  const int wave    = tid >> 5;              // 0..7
  const int lane    = tid & 31;
  const int lane_lo = lane & 15;
  const int half    = lane >> 4;
  const int mrow0   = blockIdx.x * (MT * 16);

  {
    const uint4* g = (const uint4*)(qkv + (size_t)mrow0 * D_DIM);
    for (int i = tid; i < MT * 16 * (D_DIM / 8); i += 256) {
      int r = i >> 7;
      int c = i & 127;
      *(uint4*)(smem + r * LDS_STRIDE + c * 8) = g[r * 128 + c];
    }
  }
  __syncthreads();

  const __bf16* abase = smem + lane_lo * LDS_STRIDE + half * 16;

#pragma unroll 1
  for (int nti = 0; nti < 8; ++nti) {
    const int nt   = wave * 8 + nti;         // 64 N-tiles over 8 waves
    const int colg = nt * 16 + lane_lo;
    const __bf16* wo_l = woT + (size_t)colg * D_DIM + half * 16;

    f32x8 cc[MT];
#pragma unroll
    for (int mt = 0; mt < MT; ++mt)
#pragma unroll
      for (int j = 0; j < 8; ++j) cc[mt][j] = 0.f;

    for (int ks = 0; ks < KSTEPS; ++ks) {
      const int k0 = ks * 32;
      bf16x16 bf = *(const bf16x16*)(wo_l + k0);
      bf16x16 af[MT];
#pragma unroll
      for (int mt = 0; mt < MT; ++mt)
        af[mt] = load_a_frag(abase + mt * 16 * LDS_STRIDE + k0);
#pragma unroll
      for (int mt = 0; mt < MT; ++mt)
        cc[mt] = wmma_bf16(af[mt], bf, cc[mt]);
    }

    const float bov = bo[colg];
#pragma unroll
    for (int mt = 0; mt < MT; ++mt)
#pragma unroll
      for (int j = 0; j < 8; ++j) {
        int rowg = mrow0 + mt * 16 + j + half * 8;
        out[(size_t)rowg * D_DIM + colg] = cc[mt][j] + bov;
      }
  }
}

// ---------------------------------------------------------------------------
extern "C" void kernel_launch(void* const* d_in, const int* in_sizes, int n_in,
                              void* d_out, int out_size, void* d_ws, size_t ws_size,
                              hipStream_t stream) {
  const float* x  = (const float*)d_in[0];
  const float* Wq = (const float*)d_in[1];
  const float* bq = (const float*)d_in[2];
  const float* Wk = (const float*)d_in[3];
  const float* bk = (const float*)d_in[4];
  const float* Wv = (const float*)d_in[5];
  const float* bv = (const float*)d_in[6];
  const float* Wo = (const float*)d_in[7];
  const float* bo = (const float*)d_in[8];
  float* out = (float*)d_out;

  // workspace layout (bytes): x_bf16 32MB | 4x W_T bf16 2MB | qkv_bf16 32MB
  char* ws = (char*)d_ws;
  __bf16* xb   = (__bf16*)(ws);
  __bf16* wqT  = (__bf16*)(ws + (size_t)33554432);
  __bf16* wkT  = (__bf16*)(ws + (size_t)33554432 + 2097152);
  __bf16* wvT  = (__bf16*)(ws + (size_t)33554432 + 2u * 2097152);
  __bf16* woT  = (__bf16*)(ws + (size_t)33554432 + 3u * 2097152);
  __bf16* qkvb = (__bf16*)(ws + (size_t)33554432 + 4u * 2097152);

  const int nX = ROWS * D_DIM;               // 16,777,216
  cvt_x_kernel<<<(nX + 255) / 256, 256, 0, stream>>>(x, xb, nX);

  const int nW = D_DIM * D_DIM;              // 1,048,576
  cvt_wT_kernel<<<nW / 256, 256, 0, stream>>>(Wq, wqT);
  cvt_wT_kernel<<<nW / 256, 256, 0, stream>>>(Wk, wkT);
  cvt_wT_kernel<<<nW / 256, 256, 0, stream>>>(Wv, wvT);
  cvt_wT_kernel<<<nW / 256, 256, 0, stream>>>(Wo, woT);

  const size_t ldsBytes = (size_t)(MT * 16) * LDS_STRIDE * sizeof(__bf16); // 133,120
  qkv_kernel<<<ROWS / (MT * 16), 256, ldsBytes, stream>>>(
      xb, wqT, wkT, wvT, bq, bk, bv, qkvb);

  out_kernel<<<ROWS / (MT * 16), 256, ldsBytes, stream>>>(
      qkvb, woT, bo, out);
}
